// Mamba4Rec_33432025432249
// MI455X (gfx1250) — compile-verified, matrix operand order checked by reading, not run on hardware
//
#include <hip/hip_runtime.h>
#include <hip/hip_bf16.h>
#include <math.h>
#include <stdint.h>

// ---------------- model dims ----------------
#define VOCAB 50000
#define Dm    64          // d_model
#define Nst   32          // d_state
#define Kc    4           // d_conv
#define DI    128         // d_inner
#define Rr    4           // dt_rank
#define Ff    256         // ffn inner
#define Bb    1024
#define Lseq  200
#define Mtok  (Bb * Lseq) // 204800, divisible by 16
#define EPSLN 1e-12f

typedef __attribute__((ext_vector_type(16))) __bf16   v16bf;
typedef __attribute__((ext_vector_type(8)))  float    v8f;
typedef __attribute__((ext_vector_type(4)))  uint32_t v4u;
typedef __attribute__((ext_vector_type(8)))  uint32_t v8u;

// ---------------- helpers ----------------
__device__ __forceinline__ float wave_sum32(float v) {
#pragma unroll
  for (int off = 16; off > 0; off >>= 1) v += __shfl_xor(v, off, 32);
  return v;
}

__device__ __forceinline__ float silu_f(float x) { return x / (1.0f + __expf(-x)); }

__device__ __forceinline__ float softplus_f(float x) {
  return (x > 20.0f) ? x : log1pf(__expf(x));
}

__device__ __forceinline__ float gelu_tanh(float x) {
  const float c0 = 0.7978845608028654f; // sqrt(2/pi)
  float t = tanhf(c0 * (x + 0.044715f * x * x * x));
  return 0.5f * x * (1.0f + t);
}

// ---- Tensor Data Mover: DMA a 2D tile [rows x dim0] (row stride `stride`,
// units of elements) from global into LDS offset `lds_off`, contiguous.
// dsize_code: 0=1B,1=2B,2=4B,3=8B. D# built per CDNA5 ISA.
__device__ __forceinline__ void tdm_load_2d(const void* gptr, uint32_t lds_off,
                                            uint32_t dsize_code, uint32_t dim0,
                                            uint32_t rows, uint32_t stride) {
  uint64_t ga = (uint64_t)(uintptr_t)gptr;
  v4u g0;
  g0[0] = 1u;                                   // count=1 valid descriptor
  g0[1] = lds_off;                              // lds_addr
  g0[2] = (uint32_t)(ga & 0xffffffffu);         // global_addr[31:0]
  g0[3] = (uint32_t)((ga >> 32) & 0x1ffffffu)   // global_addr[56:32]
          | (2u << 30);                         // type = 2 ("image")
  v8u g1;
  g1[0] = (dsize_code & 3u) << 16;              // workgroup_mask=0, data_size
  g1[1] = (dim0 & 0xffffu) << 16;               // tensor_dim0[15:0] @ bits[63:48]
  g1[2] = ((dim0 >> 16) & 0xffffu)              // tensor_dim0[31:16]
          | ((rows & 0xffffu) << 16);           // tensor_dim1[15:0]
  g1[3] = ((rows >> 16) & 0xffffu)              // tensor_dim1[31:16]
          | ((dim0 & 0xffffu) << 16);           // tile_dim0
  g1[4] = (rows & 0xffffu);                     // tile_dim1 (tile_dim2=0)
  g1[5] = stride;                               // tensor_dim0_stride[31:0]
  g1[6] = 0u;                                   // stride[47:32], dim1_stride lo
  g1[7] = 0u;
  asm volatile("tensor_load_to_lds %0, %1" :: "s"(g0), "s"(g1) : "memory");
}

// ---------------- weight prep ----------------
__global__ void f32_to_bf16_kernel(const float* __restrict__ src,
                                   __bf16* __restrict__ dst, int n) {
  int i = blockIdx.x * blockDim.x + threadIdx.x;
  if (i < n) dst[i] = (__bf16)src[i];
}

// W_big[192,128]: rows 0..127  = dt_proj_w @ x_proj_w[:R]
//                 rows 128..191 = x_proj_w[R : R+2N]
__global__ void build_wbig_kernel(const float* __restrict__ x_proj_w, // [68,128]
                                  const float* __restrict__ dt_proj_w,// [128,4]
                                  __bf16* __restrict__ wbig) {
  int i = blockIdx.x * blockDim.x + threadIdx.x;
  if (i >= 192 * DI) return;
  int o = i / DI, e = i % DI;
  float v;
  if (o < DI) {
    v = 0.0f;
#pragma unroll
    for (int r = 0; r < Rr; ++r)
      v += dt_proj_w[o * Rr + r] * x_proj_w[r * DI + e];
  } else {
    v = x_proj_w[(Rr + (o - DI)) * DI + e];
  }
  wbig[i] = (__bf16)v;
}

__global__ void build_aneg_kernel(const float* __restrict__ A_log,
                                  float* __restrict__ aneg) {
  int i = blockIdx.x * blockDim.x + threadIdx.x;
  if (i < DI * Nst) aneg[i] = -__expf(A_log[i]);
}

// ---------------- embedding + LayerNorm -> bf16 [M,64] ----------------
__global__ void embed_ln_kernel(const int* __restrict__ ids,
                                const float* __restrict__ emb,
                                const float* __restrict__ w,
                                const float* __restrict__ b,
                                __bf16* __restrict__ out) {
  int lane = threadIdx.x & 31;
  int tok  = blockIdx.x * (blockDim.x >> 5) + (threadIdx.x >> 5);
  if (tok >= Mtok) return;
  const float* row = emb + (size_t)ids[tok] * Dm;
  float v0 = row[lane], v1 = row[lane + 32];
  float mu = wave_sum32(v0 + v1) * (1.0f / Dm);
  float d0 = v0 - mu, d1 = v1 - mu;
  float var = wave_sum32(d0 * d0 + d1 * d1) * (1.0f / Dm);
  float rs = rsqrtf(var + EPSLN);
  out[(size_t)tok * Dm + lane]      = (__bf16)(d0 * rs * w[lane] + b[lane]);
  out[(size_t)tok * Dm + lane + 32] = (__bf16)(d1 * rs * w[lane + 32] + b[lane + 32]);
}

// ------------- WMMA GEMM, W panel staged in LDS by the TDM ---------------
// C[M, NT*16] = A[M,K] * W[NT*16, K]^T.  K in {64,128,256} templated so the
// whole K chain is unrolled; each wave owns one 16-row A panel in VGPRs and
// sweeps all NT output tiles against the LDS-resident weights.
template <int K, int NT>
__global__ void gemm_lds_kernel(const __bf16* __restrict__ A,
                                const __bf16* __restrict__ W,
                                float* __restrict__ C, int Mtiles) {
  __shared__ __bf16 sW[NT * 16 * K];
  if ((threadIdx.x >> 5) == 0) {              // wave 0 issues the tensor DMA
    tdm_load_2d(W, 0u, 1u /*2B*/, (uint32_t)K, (uint32_t)(NT * 16), (uint32_t)K);
    __builtin_amdgcn_s_wait_tensorcnt(0);
  }
  __syncthreads();

  int lane = threadIdx.x & 31;
  int wave = blockIdx.x * (blockDim.x >> 5) + (threadIdx.x >> 5);
  if (wave >= Mtiles) return;                 // wave-uniform: EXEC all-1s
  int m0 = wave << 4;

  // A lane layout: lane<16 -> row m0+lane, K chunks {0..7,16..23} (+32 per step)
  //                lane>=16 -> same row set, K chunks {8..15,24..31}
  int rsel = lane & 15;
  int kh   = (lane >> 4) * 8;
  constexpr int KS = K / 32;

  v16bf a[KS];
  {
    const __bf16* pA = A + (size_t)(m0 + rsel) * K + kh;
#pragma unroll
    for (int ks = 0; ks < KS; ++ks)
#pragma unroll
      for (int j = 0; j < 8; ++j) {
        a[ks][j]     = pA[ks * 32 + j];
        a[ks][j + 8] = pA[ks * 32 + 16 + j];
      }
  }

  int hi  = lane >> 4;
  int ldc = NT * 16;
#pragma unroll
  for (int nt = 0; nt < NT; ++nt) {
    const __bf16* pB = sW + (size_t)(nt * 16 + rsel) * K + kh;  // B = W^T column
    v8f acc = {};
#pragma unroll
    for (int ks = 0; ks < KS; ++ks) {
      v16bf bm;
#pragma unroll
      for (int j = 0; j < 8; ++j) {
        bm[j]     = pB[ks * 32 + j];
        bm[j + 8] = pB[ks * 32 + 16 + j];
      }
      acc = __builtin_amdgcn_wmma_f32_16x16x32_bf16(
          false, a[ks], false, bm, (short)0, acc, false, false);
    }
    float* pC = C + (size_t)(m0 + hi * 8) * ldc + nt * 16 + (lane & 15);
#pragma unroll
    for (int r = 0; r < 8; ++r) pC[(size_t)r * ldc] = acc[r];
  }
}

// ---------------- depthwise causal conv (K=4) + SiLU --------------------
__global__ void conv_silu_kernel(const float* __restrict__ xz,
                                 const float* __restrict__ conv_w, // [DI,1,K]
                                 const float* __restrict__ conv_b,
                                 float* __restrict__ xs,
                                 __bf16* __restrict__ xs_bf) {
  int gid = blockIdx.x * blockDim.x + threadIdx.x;
  if (gid >= Mtok * DI) return;
  int row = gid >> 7;           // b*L + t
  int d   = gid & (DI - 1);
  int t   = row % Lseq;
  float acc = conv_b[d];
#pragma unroll
  for (int j = 0; j < Kc; ++j) {
    int tt = t - (Kc - 1) + j;
    if (tt >= 0)
      acc += conv_w[d * Kc + j] * xz[(size_t)(row - t + tt) * 256 + d];
  }
  float s = silu_f(acc);
  xs[gid]    = s;
  xs_bf[gid] = (__bf16)s;
}

// -------- selective scan + D-skip + SiLU gate; B/C slab TDM'd to LDS ------
// block = 128 threads (one per channel d), grid = B. One tensor DMA pulls the
// strided [L x 64] B/C block (cols 128..191 of dbc rows) into LDS, then the
// whole 200-step recurrence runs barrier-free.
__global__ void scan_kernel(const float* __restrict__ dbc,   // [M,192]
                            const float* __restrict__ xs,    // [M,128]
                            const float* __restrict__ xz,    // [M,256] (z at 128+)
                            const float* __restrict__ dt_b,  // [128]
                            const float* __restrict__ aneg,  // [128,32]
                            const float* __restrict__ Dp,    // [128]
                            __bf16* __restrict__ yg) {       // [M,128] bf16
  __shared__ float sBC[Lseq * 2 * Nst];       // 51.2 KB
  int d = threadIdx.x;
  int b = blockIdx.x;
  if ((threadIdx.x >> 5) == 0) {
    const float* src = dbc + (size_t)b * Lseq * 192 + DI;   // first row's B ptr
    tdm_load_2d(src, 0u, 2u /*4B*/, 2 * Nst, Lseq, 192);
    __builtin_amdgcn_s_wait_tensorcnt(0);
  }
  __syncthreads();

  float Arow[Nst];
#pragma unroll
  for (int n = 0; n < Nst; ++n) Arow[n] = aneg[d * Nst + n];
  float h[Nst];
#pragma unroll
  for (int n = 0; n < Nst; ++n) h[n] = 0.0f;
  float dtb = dt_b[d], Dd = Dp[d];

  for (int t = 0; t < Lseq; ++t) {
    size_t row = (size_t)b * Lseq + t;
    const float* bc = sBC + t * 2 * Nst;
    float dt  = softplus_f(dbc[row * 192 + d] + dtb);
    float xv  = xs[row * DI + d];
    float dtx = dt * xv;
    float y = 0.0f;
#pragma unroll
    for (int n = 0; n < Nst; ++n) {
      float dA = __expf(dt * Arow[n]);
      h[n] = dA * h[n] + dtx * bc[n];
      y += h[n] * bc[Nst + n];
    }
    float z = xz[row * 256 + DI + d];
    yg[row * DI + d] = (__bf16)((y + Dd * xv) * silu_f(z));
  }
}

// ---------------- LayerNorm over D=64: f32 in -> f32 + bf16 out ----------
__global__ void ln64_kernel(const float* __restrict__ x,
                            const float* __restrict__ w,
                            const float* __restrict__ b,
                            float* __restrict__ out_f,
                            __bf16* __restrict__ out_bf) {
  int lane = threadIdx.x & 31;
  int tok  = blockIdx.x * (blockDim.x >> 5) + (threadIdx.x >> 5);
  if (tok >= Mtok) return;
  const float* row = x + (size_t)tok * Dm;
  float v0 = row[lane], v1 = row[lane + 32];
  float mu = wave_sum32(v0 + v1) * (1.0f / Dm);
  float d0 = v0 - mu, d1 = v1 - mu;
  float var = wave_sum32(d0 * d0 + d1 * d1) * (1.0f / Dm);
  float rs = rsqrtf(var + EPSLN);
  float o0 = d0 * rs * w[lane] + b[lane];
  float o1 = d1 * rs * w[lane + 32] + b[lane + 32];
  out_f[(size_t)tok * Dm + lane]       = o0;
  out_f[(size_t)tok * Dm + lane + 32]  = o1;
  out_bf[(size_t)tok * Dm + lane]      = (__bf16)o0;
  out_bf[(size_t)tok * Dm + lane + 32] = (__bf16)o1;
}

// ---------------- bias + GELU(tanh) -> bf16 [M,256] ----------------------
__global__ void bias_gelu_kernel(const float* __restrict__ f1,
                                 const float* __restrict__ b1,
                                 __bf16* __restrict__ out) {
  int i = blockIdx.x * blockDim.x + threadIdx.x;
  if (i >= Mtok * Ff) return;
  out[i] = (__bf16)gelu_tanh(f1[i] + b1[i & (Ff - 1)]);
}

// ---------------- final: LN(f2 + b2 + hid) -> d_out f32 ------------------
__global__ void final_ln_kernel(const float* __restrict__ f2,
                                const float* __restrict__ b2,
                                const float* __restrict__ hid,
                                const float* __restrict__ w,
                                const float* __restrict__ b,
                                float* __restrict__ out) {
  int lane = threadIdx.x & 31;
  int tok  = blockIdx.x * (blockDim.x >> 5) + (threadIdx.x >> 5);
  if (tok >= Mtok) return;
  size_t base = (size_t)tok * Dm;
  float v0 = f2[base + lane]      + b2[lane]      + hid[base + lane];
  float v1 = f2[base + lane + 32] + b2[lane + 32] + hid[base + lane + 32];
  float mu = wave_sum32(v0 + v1) * (1.0f / Dm);
  float d0 = v0 - mu, d1 = v1 - mu;
  float var = wave_sum32(d0 * d0 + d1 * d1) * (1.0f / Dm);
  float rs = rsqrtf(var + EPSLN);
  out[base + lane]      = d0 * rs * w[lane] + b[lane];
  out[base + lane + 32] = d1 * rs * w[lane + 32] + b[lane + 32];
}

// ======================= host orchestration =============================
extern "C" void kernel_launch(void* const* d_in, const int* in_sizes, int n_in,
                              void* d_out, int out_size, void* d_ws, size_t ws_size,
                              hipStream_t stream) {
  const int*   ids       = (const int*)  d_in[0];
  const float* emb       = (const float*)d_in[1];
  const float* ln_w      = (const float*)d_in[2];
  const float* ln_b      = (const float*)d_in[3];
  const float* in_proj_w = (const float*)d_in[4];
  const float* conv_w    = (const float*)d_in[5];
  const float* conv_b    = (const float*)d_in[6];
  const float* x_proj_w  = (const float*)d_in[7];
  const float* dt_proj_w = (const float*)d_in[8];
  const float* dt_proj_b = (const float*)d_in[9];
  const float* A_log     = (const float*)d_in[10];
  const float* D_param   = (const float*)d_in[11];
  const float* out_proj_w= (const float*)d_in[12];
  const float* mln_w     = (const float*)d_in[13];
  const float* mln_b     = (const float*)d_in[14];
  const float* ffn_w1    = (const float*)d_in[15];
  const float* ffn_b1    = (const float*)d_in[16];
  const float* ffn_w2    = (const float*)d_in[17];
  const float* ffn_b2    = (const float*)d_in[18];
  const float* fln_w     = (const float*)d_in[19];
  const float* fln_b     = (const float*)d_in[20];

  const size_t M = (size_t)Mtok;
  size_t off = 0;
  auto alloc = [&](size_t bytes) {
    size_t o = off;
    off = (off + bytes + 255) & ~(size_t)255;
    return o;
  };
  size_t oWIn  = alloc(256 * 64 * sizeof(__bf16));
  size_t oWBig = alloc(192 * DI * sizeof(__bf16));
  size_t oWOut = alloc(64 * DI * sizeof(__bf16));
  size_t oWF1  = alloc(Ff * Dm * sizeof(__bf16));
  size_t oWF2  = alloc(Dm * Ff * sizeof(__bf16));
  size_t oAneg = alloc(DI * Nst * sizeof(float));
  size_t oXBf  = alloc(M * Dm * sizeof(__bf16));
  size_t oXZ   = alloc(M * 256 * sizeof(float));    // reused later for f1
  size_t oXS   = alloc(M * DI * sizeof(float));     // reused later for f2
  size_t oXSBf = alloc(M * DI * sizeof(__bf16));
  size_t oDBC  = alloc(M * 192 * sizeof(float));    // reused later for f_bf16
  size_t oYGBf = alloc(M * DI * sizeof(__bf16));
  size_t oMout = alloc(M * Dm * sizeof(float));
  size_t oHid  = alloc(M * Dm * sizeof(float));
  size_t oHidB = alloc(M * Dm * sizeof(__bf16));
  if (off > ws_size) return;  // workspace too small; deterministic no-op

  char* ws = (char*)d_ws;
  __bf16* wIn   = (__bf16*)(ws + oWIn);
  __bf16* wBig  = (__bf16*)(ws + oWBig);
  __bf16* wOut  = (__bf16*)(ws + oWOut);
  __bf16* wF1   = (__bf16*)(ws + oWF1);
  __bf16* wF2   = (__bf16*)(ws + oWF2);
  float*  aneg  = (float*) (ws + oAneg);
  __bf16* xBf   = (__bf16*)(ws + oXBf);
  float*  xz    = (float*) (ws + oXZ);
  float*  xs    = (float*) (ws + oXS);
  __bf16* xsBf  = (__bf16*)(ws + oXSBf);
  float*  dbc   = (float*) (ws + oDBC);
  __bf16* ygBf  = (__bf16*)(ws + oYGBf);
  float*  mout  = (float*) (ws + oMout);
  float*  hid   = (float*) (ws + oHid);
  __bf16* hidBf = (__bf16*)(ws + oHidB);
  float*  f1    = xz;                      // reuse (xz dead after scan)
  __bf16* fBf   = (__bf16*)(ws + oDBC);    // reuse (dbc dead after scan)
  float*  f2    = xs;                      // reuse (xs dead after scan)

  // --- weight prep ---
  f32_to_bf16_kernel<<<(256*64 + 255) / 256, 256, 0, stream>>>(in_proj_w, wIn, 256*64);
  f32_to_bf16_kernel<<<(64*DI + 255) / 256, 256, 0, stream>>>(out_proj_w, wOut, 64*DI);
  f32_to_bf16_kernel<<<(Ff*Dm + 255) / 256, 256, 0, stream>>>(ffn_w1, wF1, Ff*Dm);
  f32_to_bf16_kernel<<<(Dm*Ff + 255) / 256, 256, 0, stream>>>(ffn_w2, wF2, Dm*Ff);
  build_wbig_kernel<<<(192*DI + 255) / 256, 256, 0, stream>>>(x_proj_w, dt_proj_w, wBig);
  build_aneg_kernel<<<(DI*Nst + 255) / 256, 256, 0, stream>>>(A_log, aneg);

  // --- embedding + LN ---
  embed_ln_kernel<<<Mtok / 4, 128, 0, stream>>>(ids, emb, ln_w, ln_b, xBf);

  const int mT = Mtok / 16;              // 12800 M-tiles
  const int gBlocks = mT / 8;            // 8 waves (M-tiles) per 256-thread block

  // --- in_proj: [M,64] x [256,64]^T -> xz [M,256] ---
  gemm_lds_kernel<64, 16><<<gBlocks, 256, 0, stream>>>(xBf, wIn, xz, mT);

  // --- depthwise causal conv + SiLU -> xs ---
  conv_silu_kernel<<<(Mtok * DI + 255) / 256, 256, 0, stream>>>(xz, conv_w, conv_b, xs, xsBf);

  // --- fused x_proj/dt_proj: [M,128] x [192,128]^T -> dbc [M,192] ---
  gemm_lds_kernel<128, 12><<<gBlocks, 256, 0, stream>>>(xsBf, wBig, dbc, mT);

  // --- selective scan + D-skip + gate -> yg bf16 [M,128] ---
  scan_kernel<<<Bb, DI, 0, stream>>>(dbc, xs, xz, dt_proj_b, aneg, D_param, ygBf);

  // --- out_proj: [M,128] x [64,128]^T -> mout [M,64] ---
  gemm_lds_kernel<128, 4><<<gBlocks, 256, 0, stream>>>(ygBf, wOut, mout, mT);

  // --- post-mamba LN -> hid ---
  ln64_kernel<<<Mtok / 4, 128, 0, stream>>>(mout, mln_w, mln_b, hid, hidBf);

  // --- FFN1: [M,64] x [256,64]^T -> f1 [M,256] ; bias+GELU -> bf16 ---
  gemm_lds_kernel<64, 16><<<gBlocks, 256, 0, stream>>>(hidBf, wF1, f1, mT);
  bias_gelu_kernel<<<(Mtok * Ff + 255) / 256, 256, 0, stream>>>(f1, ffn_b1, fBf);

  // --- FFN2: [M,256] x [64,256]^T -> f2 [M,64] ---
  gemm_lds_kernel<256, 4><<<gBlocks, 256, 0, stream>>>(fBf, wF2, f2, mT);

  // --- final residual LN -> d_out ---
  final_ln_kernel<<<Mtok / 4, 128, 0, stream>>>(f2, ffn_b2, hid, fln_w, fln_b, (float*)d_out);
}
